// LocalAttention_16767552323852
// MI455X (gfx1250) — matrix operand order repaired and checked by reading, not compiled, not run on hardware
//
#include <hip/hip_runtime.h>

// ---------------------------------------------------------------------------
// Local windowed attention with rotary embeddings, fused single kernel.
// One workgroup (256 threads = 8 wave32) per window; all GEMMs via
// v_wmma_f32_16x16x32_f16 (f16 operands, f32 accumulate).
// ---------------------------------------------------------------------------

typedef __attribute__((ext_vector_type(16))) _Float16 v16h;
typedef __attribute__((ext_vector_type(8)))  float    v8f;
typedef __attribute__((ext_vector_type(4)))  unsigned int v4u;

namespace {

constexpr int kB   = 4;
constexpr int kT   = 8192;
constexpr int kC   = 128;
constexpr int kH   = 8;
constexpr int kW   = 64;     // query window
constexpr int kCtx = 128;    // context window = 2W
constexpr int kDH  = 16;
constexpr int kNW  = kT / kW;        // 128 windows per batch row
constexpr int kNWIN = kB * kNW;      // 512 workgroups
constexpr int kTB  = 256;            // threads per block (8 waves)

// ---- LDS layout (byte offsets, all 128-aligned). Buffers alias across phases.
constexpr int XQ_OFF   = 0;          // f16 [64][128]   (later: CTX f16 [64][128])
constexpr int XK_OFF   = 16384;      // f16 [128][128]  (later: SBUF f32 [64][128])
constexpr int WF_OFF   = 49152;      // f16 [128][128]  current weight matrix
constexpr int QR_OFF   = 81920;      // f16 [64][128]   rotated Q
constexpr int KR_OFF   = 98304;      // f16 [128][128]  rotated K
constexpr int VT_OFF   = 131072;     // f16 [128][128]  V transposed: [feat][key]
constexpr int TRIG_OFF = 163840;     // f32 cq[1024] sq[1024] ck[2048] sk[2048]
                                     // (later: P f16 [64][128])
constexpr int ZERO_OFF = 188416;     // f16 [64] zeros (K-padding source)
constexpr int KM_OFF   = 188544;     // int [128] key-valid mask
constexpr int SMEM_BYTES = 189056;

union H16U { v16h v; v4u q[2]; };

__device__ inline v16h ld_pair(const _Float16* p0, const _Float16* p1) {
  H16U u;
  u.q[0] = *(const v4u*)p0;
  u.q[1] = *(const v4u*)p1;
  return u.v;
}

__device__ inline v8f wmma_f16(v16h a, v16h b, v8f c) {
  return __builtin_amdgcn_wmma_f32_16x16x32_f16(false, a, false, b,
                                                (short)0, c, false, false);
}

} // namespace

__global__ __launch_bounds__(kTB)
void local_attn_rotary_kernel(const float* __restrict__ x,
                              const float* __restrict__ cosp,
                              const float* __restrict__ sinp,
                              const float* __restrict__ Wq, const float* __restrict__ bq,
                              const float* __restrict__ Wk, const float* __restrict__ bk,
                              const float* __restrict__ Wv, const float* __restrict__ bv,
                              const float* __restrict__ Wo, const float* __restrict__ bo,
                              float* __restrict__ out)
{
  __shared__ __align__(16) char smem[SMEM_BYTES];

  _Float16* XQ   = (_Float16*)(smem + XQ_OFF);
  _Float16* CTX  = (_Float16*)(smem + XQ_OFF);     // alias (live after Q used)
  _Float16* XK   = (_Float16*)(smem + XK_OFF);
  float*    SBUF = (float*)   (smem + XK_OFF);     // alias (live after V proj)
  _Float16* WF   = (_Float16*)(smem + WF_OFF);
  _Float16* QR   = (_Float16*)(smem + QR_OFF);
  _Float16* KR   = (_Float16*)(smem + KR_OFF);
  _Float16* VT   = (_Float16*)(smem + VT_OFF);
  float*    CQ   = (float*)   (smem + TRIG_OFF);
  float*    SQ   = CQ + kW * kDH;                  // 1024
  float*    CK   = SQ + kW * kDH;
  float*    SK   = CK + kCtx * kDH;
  _Float16* Pb   = (_Float16*)(smem + TRIG_OFF);   // alias (live after K proj)
  _Float16* ZEROh= (_Float16*)(smem + ZERO_OFF);
  int*      KM   = (int*)     (smem + KM_OFF);

  const int tid  = threadIdx.x;
  const int wave = tid >> 5;
  const int lane = tid & 31;
  const int l16  = lane & 15;
  const int hi   = lane >> 4;                      // 0 | 1
  const int n    = blockIdx.x;
  const int b    = n / kNW;
  const int w    = n % kNW;
  const int q0   = w * kW;                         // first query position
  const long xbase = (long)b * kT * kC;

  // ---------------- phase 0: stage inputs into LDS ----------------
  if (tid < 64) ZEROh[tid] = (_Float16)0.f;
  if (tid < kCtx) {
    int t = q0 - 32 + tid;
    KM[tid] = (t >= 0 && t < kT) ? 1 : 0;
  }
  for (int i = tid; i < kW * kC; i += kTB) {
    int m = i >> 7, c = i & 127;
    XQ[i] = (_Float16)x[xbase + (long)(q0 + m) * kC + c];
  }
  for (int i = tid; i < kCtx * kC; i += kTB) {
    int j = i >> 7, c = i & 127;
    int t = q0 - 32 + j;
    float v = (t >= 0 && t < kT) ? x[xbase + (long)t * kC + c] : 0.f;
    XK[i] = (_Float16)v;
  }
  for (int i = tid; i < kW * kDH; i += kTB) {
    int m = i >> 4, d = i & 15;
    CQ[i] = cosp[(q0 + m) * kDH + d];
    SQ[i] = sinp[(q0 + m) * kDH + d];
  }
  for (int i = tid; i < kCtx * kDH; i += kTB) {
    int j = i >> 4, d = i & 15;
    int t = q0 - 32 + j;
    float cv = 0.f, sv = 0.f;
    if (t >= 0 && t < kT) { cv = cosp[t * kDH + d]; sv = sinp[t * kDH + d]; }
    CK[i] = cv; SK[i] = sv;
  }
  __syncthreads();

  auto load_w = [&](const float* Wsrc) {
    for (int i = tid; i < kC * kC; i += kTB) WF[i] = (_Float16)Wsrc[i];
  };

  // ---------------- phase 1: Q = XQ * Wq^T, rotary -> QR ----------------
  load_w(Wq);
  __syncthreads();
  for (int i = 0; i < 4; i++) {                    // 32 tiles / 8 waves
    int tt = wave * 4 + i;
    int mt = tt >> 3, nt = tt & 7;
    v8f acc = {};
    #pragma unroll
    for (int ks = 0; ks < 4; ks++) {
      int k0 = ks * 32;
      int arow = mt * 16 + l16;
      int akb  = k0 + hi * 8;
      v16h a = ld_pair(XQ + arow * kC + akb, XQ + arow * kC + akb + 16);
      int brow = nt * 16 + l16;
      int bkb  = k0 + hi * 8;
      v16h bm = ld_pair(WF + brow * kC + bkb, WF + brow * kC + bkb + 16);
      acc = wmma_f16(a, bm, acc);
    }
    int f = nt * 16 + l16;                         // head nt, dim l16
    int d = l16;
    float bias = bq[f];
    #pragma unroll
    for (int r = 0; r < 8; r++) {
      int m = mt * 16 + r + hi * 8;
      float v = acc[r] + bias;
      float p = __shfl_xor(v, 8, 32);              // rotate-half partner (d^8)
      float rot = v * CQ[m * kDH + d] + ((d < 8) ? -p : p) * SQ[m * kDH + d];
      QR[m * kC + f] = (_Float16)rot;
    }
  }
  __syncthreads();

  // ---------------- phase 2: K = XK * Wk^T, rotary -> KR ----------------
  load_w(Wk);
  __syncthreads();
  for (int i = 0; i < 8; i++) {                    // 64 tiles / 8 waves
    int tt = wave * 8 + i;
    int mt = tt >> 3, nt = tt & 7;
    v8f acc = {};
    #pragma unroll
    for (int ks = 0; ks < 4; ks++) {
      int k0 = ks * 32;
      int arow = mt * 16 + l16;
      int akb  = k0 + hi * 8;
      v16h a = ld_pair(XK + arow * kC + akb, XK + arow * kC + akb + 16);
      int brow = nt * 16 + l16;
      int bkb  = k0 + hi * 8;
      v16h bm = ld_pair(WF + brow * kC + bkb, WF + brow * kC + bkb + 16);
      acc = wmma_f16(a, bm, acc);
    }
    int f = nt * 16 + l16;
    int d = l16;
    float bias = bk[f];
    #pragma unroll
    for (int r = 0; r < 8; r++) {
      int j = mt * 16 + r + hi * 8;                // key index 0..127
      float v = acc[r] + bias;
      float p = __shfl_xor(v, 8, 32);
      float rot = v * CK[j * kDH + d] + ((d < 8) ? -p : p) * SK[j * kDH + d];
      KR[j * kC + f] = (_Float16)rot;
    }
  }
  __syncthreads();

  // ---------------- phase 3: V = XK * Wv^T -> VT [feat][key] ----------------
  load_w(Wv);
  __syncthreads();
  for (int i = 0; i < 8; i++) {
    int tt = wave * 8 + i;
    int mt = tt >> 3, nt = tt & 7;
    v8f acc = {};
    #pragma unroll
    for (int ks = 0; ks < 4; ks++) {
      int k0 = ks * 32;
      int arow = mt * 16 + l16;
      int akb  = k0 + hi * 8;
      v16h a = ld_pair(XK + arow * kC + akb, XK + arow * kC + akb + 16);
      int brow = nt * 16 + l16;
      int bkb  = k0 + hi * 8;
      v16h bm = ld_pair(WF + brow * kC + bkb, WF + brow * kC + bkb + 16);
      acc = wmma_f16(a, bm, acc);
    }
    int f = nt * 16 + l16;
    float bias = bv[f];
    #pragma unroll
    for (int r = 0; r < 8; r++) {
      int j = mt * 16 + r + hi * 8;
      VT[f * kCtx + j] = (_Float16)(acc[r] + bias); // transposed store
    }
  }
  __syncthreads();

  // ---------------- phase 4: attention, one head at a time ----------------
  for (int h = 0; h < kH; h++) {
    // scores S = Q_h (64x16, K-padded to 32) x K_h^T -> SBUF f32 [64][128]
    for (int i = 0; i < 4; i++) {
      int tt = wave * 4 + i;
      int mt = tt >> 3, nt = tt & 7;
      int arow = mt * 16 + l16;
      int akb  = h * 16 + hi * 8;                  // real K 0..15; upper = zeros
      v16h a = ld_pair(QR + arow * kC + akb, ZEROh);
      const _Float16* bp0 = KR + (nt * 16 + l16) * kC + h * 16 + hi * 8;
      const _Float16* bp1 = ZEROh;
      v16h bm = ld_pair(bp0, bp1);
      v8f acc = {};
      acc = wmma_f16(a, bm, acc);
      #pragma unroll
      for (int r = 0; r < 8; r++) {
        int m = mt * 16 + r + hi * 8;
        SBUF[m * kCtx + nt * 16 + l16] = acc[r];
      }
    }
    __syncthreads();

    // softmax over 128 keys, one row per thread (skewed to spread LDS banks)
    if (tid < kW) {
      int base = tid * kCtx;
      int skew = (tid * 2) & 127;
      float mx = -3.4e38f;
      for (int jj = 0; jj < kCtx; jj++) {
        int j = (jj + skew) & 127;
        float s = KM[j] ? SBUF[base + j] * 0.25f : -1e9f;
        mx = fmaxf(mx, s);
      }
      float sum = 0.f;
      for (int jj = 0; jj < kCtx; jj++) {
        int j = (jj + skew) & 127;
        float e = KM[j] ? __expf(SBUF[base + j] * 0.25f - mx) : 0.f;
        sum += e;
        Pb[base + j] = (_Float16)e;                // e <= 1, safe in f16
      }
      float inv = 1.f / sum;
      for (int jj = 0; jj < kCtx; jj++) {
        int j = (jj + skew) & 127;
        Pb[base + j] = (_Float16)((float)Pb[base + j] * inv);
      }
    }
    __syncthreads();

    // out_h = P (64x128) x V_h (128x16) -> CTX columns of head h
    if (wave < 4) {
      int mt = wave;
      v8f acc = {};
      #pragma unroll
      for (int ks = 0; ks < 4; ks++) {
        int k0 = ks * 32;
        int arow = mt * 16 + l16;
        int akb  = k0 + hi * 8;
        v16h a = ld_pair(Pb + arow * kCtx + akb, Pb + arow * kCtx + akb + 16);
        int brow = h * 16 + l16;                   // VT row = feature
        int bkb  = k0 + hi * 8;
        v16h bm = ld_pair(VT + brow * kCtx + bkb, VT + brow * kCtx + bkb + 16);
        acc = wmma_f16(a, bm, acc);
      }
      #pragma unroll
      for (int r = 0; r < 8; r++) {
        int m = mt * 16 + r + hi * 8;
        CTX[m * kC + h * 16 + l16] = (_Float16)acc[r];
      }
    }
    __syncthreads();
  }

  // ---------------- phase 5: OUT = CTX * Wo^T + bo -> global ----------------
  load_w(Wo);
  __syncthreads();
  for (int i = 0; i < 4; i++) {
    int tt = wave * 4 + i;
    int mt = tt >> 3, nt = tt & 7;
    v8f acc = {};
    #pragma unroll
    for (int ks = 0; ks < 4; ks++) {
      int k0 = ks * 32;
      int arow = mt * 16 + l16;
      int akb  = k0 + hi * 8;
      v16h a = ld_pair(CTX + arow * kC + akb, CTX + arow * kC + akb + 16);
      int brow = nt * 16 + l16;
      int bkb  = k0 + hi * 8;
      v16h bm = ld_pair(WF + brow * kC + bkb, WF + brow * kC + bkb + 16);
      acc = wmma_f16(a, bm, acc);
    }
    int f = nt * 16 + l16;
    float bias = bo[f];
    #pragma unroll
    for (int r = 0; r < 8; r++) {
      int m = mt * 16 + r + hi * 8;
      out[xbase + (long)(q0 + m) * kC + f] = acc[r] + bias;
    }
  }
}

extern "C" void kernel_launch(void* const* d_in, const int* in_sizes, int n_in,
                              void* d_out, int out_size, void* d_ws, size_t ws_size,
                              hipStream_t stream) {
  (void)in_sizes; (void)n_in; (void)out_size; (void)d_ws; (void)ws_size;
  const float* x    = (const float*)d_in[0];
  // d_in[1] = padding_mask (all True; boundary masking derived geometrically)
  const float* cosp = (const float*)d_in[2];
  const float* sinp = (const float*)d_in[3];
  const float* Wq   = (const float*)d_in[4];
  const float* bq   = (const float*)d_in[5];
  const float* Wk   = (const float*)d_in[6];
  const float* bk   = (const float*)d_in[7];
  const float* Wv   = (const float*)d_in[8];
  const float* bv   = (const float*)d_in[9];
  const float* Wo   = (const float*)d_in[10];
  const float* bo   = (const float*)d_in[11];
  float* out = (float*)d_out;

  local_attn_rotary_kernel<<<kNWIN, kTB, 0, stream>>>(
      x, cosp, sinp, Wq, bq, Wk, bk, Wv, bv, Wo, bo, out);
}
